// pLSTM_60275571032446
// MI455X (gfx1250) — matrix operand order repaired and checked by reading, not compiled, not run on hardware
//
#include <hip/hip_runtime.h>

// ---------------- problem constants (match reference) ----------------
#define BB 64
#define TT 512
#define FF 512
#define HH 1024
#define H3 3072          // 3*HH, gates packed [r | c | o]
#define EPS 0.001f

// persistent recurrence configuration
#define NBLK 32                         // persistent blocks; block j owns h-cols [32j,32j+32)
#define NCOL 32                         // h-columns per block
#define GC   96                         // gate columns per block (3 * NCOL)
#define LDSK (HH + 8)                   // padded LDS row (bf16 elems) to spread banks
#define W_LDS_BYTES (GC * LDSK * 2)     // 198144 B: block's weight slice, bf16
#define LDS_BYTES (W_LDS_BYTES + BB * GC * 4)   // + 24576 B G staging = 222720 B

// ---------------- WMMA fragment types (gfx1250, wave32) ----------------
typedef __attribute__((ext_vector_type(16))) __bf16 v16bf;
typedef __attribute__((ext_vector_type(8)))  __bf16 v8bf;
typedef __attribute__((ext_vector_type(8)))  float  v8f;

// A-matrix 16x32 bf16 fragment (ISA 7.12.2): lane L holds row (L&15);
// lanes 0-15: elems 0..7 = K[0..7],  8..15 = K[16..23]
// lanes16-31: elems 0..7 = K[8..15], 8..15 = K[24..31]
static __device__ inline v16bf load_afrag(const __bf16* __restrict__ A, int lda, int lane) {
    int r  = lane & 15;
    int kh = (lane >> 4) << 3;              // 0 or 8
    const __bf16* p = A + (size_t)r * lda + kh;
    v8bf lo = *(const v8bf*)(p);            // K kh..kh+7     (16B)
    v8bf hi = *(const v8bf*)(p + 16);       // K kh+16..kh+23 (16B)
    return __builtin_shufflevector(lo, hi, 0,1,2,3,4,5,6,7,8,9,10,11,12,13,14,15);
}

// B-matrix 32x16 bf16 fragment, B stored PRE-TRANSPOSED (row n = column n of B):
// lanes 0-15: elems 0..7 = K[0..7], 8..15 = K[16..23]; lanes 16-31: K[8..15], K[24..31]
static __device__ inline v16bf load_bfrag(const __bf16* __restrict__ Bt, int ldb, int lane) {
    int n  = lane & 15;
    int ks = (lane >> 4) << 3;              // 0 or 8
    const __bf16* p = Bt + (size_t)n * ldb + ks;
    v8bf lo = *(const v8bf*)(p);
    v8bf hi = *(const v8bf*)(p + 16);
    return __builtin_shufflevector(lo, hi, 0,1,2,3,4,5,6,7,8,9,10,11,12,13,14,15);
}

static __device__ inline v8f wmma_bf16(v16bf a, v16bf b, v8f c) {
    return __builtin_amdgcn_wmma_f32_16x16x32_bf16(false, a, false, b, (short)0, c,
                                                   false, false);
}

// ---------------- setup kernels ----------------
__global__ void cast_f32_to_bf16(const float* __restrict__ in, __bf16* __restrict__ out, long n) {
    long i = blockIdx.x * (long)blockDim.x + threadIdx.x;
    if (i < n) out[i] = (__bf16)in[i];
}

// src [K,N] fp32 row-major  ->  dst [N,K] bf16 row-major (transposed cast)
__global__ void transpose_cast(const float* __restrict__ src, __bf16* __restrict__ dst,
                               int K, int N) {
    long i = blockIdx.x * (long)blockDim.x + threadIdx.x;
    if (i >= (long)K * N) return;
    int n = (int)(i / K), k = (int)(i % K);
    dst[i] = (__bf16)src[(size_t)k * N + n];
}

__global__ void init_rec(__bf16* hb, unsigned* cnt) {
    int i = blockIdx.x * blockDim.x + threadIdx.x;
    if (i < BB * HH) hb[i] = (__bf16)0.f;
    if (i == 0) *cnt = 0u;
}

// ---------------- GEMM 1: hoisted input projections ----------------
// X_bf [B*T, F] x Ut [3H, F]^T  -> xg [T, B, 3H] fp32 (time-major for the scan)
// wave = 2 M-tiles x 4 N-tiles (32x64 output), 8 wmma per K-step
__global__ __launch_bounds__(256) void gemm_xproj(const __bf16* __restrict__ X,
                                                  const __bf16* __restrict__ Ut,
                                                  float* __restrict__ xg) {
    const int lane  = threadIdx.x & 31;
    const int wave  = blockIdx.x * (blockDim.x >> 5) + (threadIdx.x >> 5);
    const int nstr  = H3 / 64;                 // 48 N-strips
    const int mstr  = (BB * TT) / 32;          // 1024 M-strips
    if (wave >= nstr * mstr) return;
    const int m0 = (wave / nstr) * 32;
    const int n0 = (wave % nstr) * 64;

    v8f acc[2][4] = {};
    for (int k0 = 0; k0 < FF; k0 += 32) {
        v16bf a[2], b[4];
#pragma unroll
        for (int mi = 0; mi < 2; ++mi)
            a[mi] = load_afrag(X + (size_t)(m0 + 16 * mi) * FF + k0, FF, lane);
#pragma unroll
        for (int ni = 0; ni < 4; ++ni)
            b[ni] = load_bfrag(Ut + (size_t)(n0 + 16 * ni) * FF + k0, FF, lane);
#pragma unroll
        for (int mi = 0; mi < 2; ++mi)
#pragma unroll
            for (int ni = 0; ni < 4; ++ni)
                acc[mi][ni] = wmma_bf16(a[mi], b[ni], acc[mi][ni]);
    }
    // C/D layout: VGPR v -> row v + 8*(lane>=16), col lane&15
    const int rofs = (lane >> 4) << 3;
    const int col0 = lane & 15;
#pragma unroll
    for (int mi = 0; mi < 2; ++mi)
#pragma unroll
        for (int ni = 0; ni < 4; ++ni)
#pragma unroll
            for (int v = 0; v < 8; ++v) {
                int m = m0 + 16 * mi + rofs + v;       // m = b*T + t
                int bidx = m >> 9, t = m & (TT - 1);
                int col = n0 + 16 * ni + col0;
                xg[((size_t)t * BB + bidx) * H3 + col] = acc[mi][ni][v];
            }
}

// ---------------- persistent recurrence kernel ----------------
// One launch runs the entire T=512 scan. Block j owns h-columns [32j, 32j+32)
// and gate columns {n, HH+n, 2*HH+n : n in slice}.  Its 96x1024 bf16 weight
// slice lives in LDS for the whole scan; c/k states live in registers; only
// h crosses blocks (via L2, ordered by one device barrier per step).
__global__ __launch_bounds__(256) void plstm_recurrence(
    __bf16* __restrict__ h_bf,            // [B, H] recurrent state, bf16 (pre-zeroed)
    const float* __restrict__ xg,         // [T, B, 3H] input projections
    const float* __restrict__ tseq,       // [B, T]
    const float* __restrict__ W_r, const float* __restrict__ W_c,
    const float* __restrict__ W_o,
    const float* __restrict__ br, const float* __restrict__ bc,
    const float* __restrict__ bo,
    float* __restrict__ out_hs,           // [B, T, H]
    float* __restrict__ out_hf,           // [B, H]
    float* __restrict__ out_cf,           // [B, H]
    unsigned* __restrict__ bar_cnt) {
    extern __shared__ char smem[];
    __bf16* Wl = (__bf16*)smem;                        // [GC][LDSK] bf16 (transposed W)
    float*  Gb = (float*)(smem + W_LDS_BYTES);         // [BB][GC] f32 staging

    const int tid  = threadIdx.x;
    const int lane = tid & 31;
    const int wv   = tid >> 5;

    // ---- one-time: stage this block's weight slice into LDS (transpose+cast) ----
    // LDS row = g*NCOL + n  <->  column (g*HH + 32*blockIdx.x + n) of [Wr|Wc|Wo]
    for (int idx = tid; idx < GC * HH; idx += 256) {
        int row = idx % GC;                 // consecutive tids -> consecutive rows
        int k   = idx / GC;
        int g   = row >> 5, n = row & (NCOL - 1);
        const float* W = (g == 0) ? W_r : (g == 1) ? W_c : W_o;
        Wl[(size_t)row * LDSK + k] = (__bf16)W[(size_t)k * HH + NCOL * blockIdx.x + n];
    }

    // ---- per-thread recurrent state: 8 (b, n) pairs ----
    const int nl   = tid & (NCOL - 1);            // local h-column
    const int bb0  = (tid >> 5) * 8;              // batch base
    const int gcol = NCOL * blockIdx.x + nl;      // global h-column
    const float vbr = br[gcol], vbc = bc[gcol], vbo = bo[gcol];
    float creg[8], kreg[8];
#pragma unroll
    for (int p = 0; p < 8; ++p) { creg[p] = 0.f; kreg[p] = 0.f; }

    // GEMM tile assignment: wave = (m-tile 0..3) x (n-group 0..1 of 3 n-tiles)
    const int mt = wv & 3;
    const int ng = wv >> 2;
    const int rofs = (lane >> 4) << 3;
    const int col0 = lane & 15;

    __syncthreads();                       // LDS weights ready

    unsigned target = 0;
    for (int t = 0; t < TT; ++t) {
        // ---- phase A: G(:, block cols) = h @ W_slice  (WMMA, B from LDS) ----
        v8f acc[3] = {};
        for (int k0 = 0; k0 < HH; k0 += 32) {
            v16bf a = load_afrag(h_bf + (size_t)(16 * mt) * HH + k0, HH, lane);
#pragma unroll
            for (int ni = 0; ni < 3; ++ni) {
                v16bf b = load_bfrag(Wl + (size_t)(16 * (3 * ng + ni)) * LDSK + k0,
                                     LDSK, lane);
                acc[ni] = wmma_bf16(a, b, acc[ni]);
            }
        }
#pragma unroll
        for (int ni = 0; ni < 3; ++ni)
#pragma unroll
            for (int v = 0; v < 8; ++v)
                Gb[(size_t)(16 * mt + rofs + v) * GC + 16 * (3 * ng + ni) + col0] =
                    acc[ni][v];
        __syncthreads();                   // G staged for this block's columns

        // ---- phase B: block-local gate math, states in registers ----
        const float* xgt = xg + (size_t)t * BB * H3;
#pragma unroll
        for (int p = 0; p < 8; ++p) {
            int b = bb0 + p;
            float tt = tseq[(size_t)b * TT + t];
            float pr = Gb[(size_t)b * GC + nl]             + xgt[(size_t)b * H3 + gcol]          + vbr;
            float pc = Gb[(size_t)b * GC + NCOL + nl]      + xgt[(size_t)b * H3 + HH + gcol]     + vbc;
            float po = Gb[(size_t)b * GC + 2 * NCOL + nl]  + xgt[(size_t)b * H3 + 2 * HH + gcol] + vbo;
            float k = kreg[p], c = creg[p];
            float r = 1.f / (1.f + __expf(-pr));
            k = r * tt + (1.f - r) * k;
            float ctil = tanhf(pc);
            float d = tt - k;
            float f = __fsqrt_rn((d + EPS) / (d + 1.0f));   // ((d+1)/(d+eps))^(-1/2)
            c = f * c + (1.f - f) * ctil;
            float o = 1.f / (1.f + __expf(-po));
            float h = o * tanhf(c);
            kreg[p] = k; creg[p] = c;
            h_bf[(size_t)b * HH + gcol] = (__bf16)h;
            out_hs[((size_t)b * TT + t) * HH + gcol] = h;
            if (t == TT - 1) {
                out_hf[(size_t)b * HH + gcol] = h;
                out_cf[(size_t)b * HH + gcol] = c;
            }
        }

        // ---- device-wide barrier: publish h, separate steps ----
        target += (unsigned)gridDim.x;
        __syncthreads();
        if (tid == 0) {
            __threadfence();
            __hip_atomic_fetch_add(bar_cnt, 1u, __ATOMIC_ACQ_REL,
                                   __HIP_MEMORY_SCOPE_AGENT);
            while (__hip_atomic_load(bar_cnt, __ATOMIC_ACQUIRE,
                                     __HIP_MEMORY_SCOPE_AGENT) < target)
                __builtin_amdgcn_s_sleep(1);
        }
        __syncthreads();
    }
}

// ---------------- launch ----------------
extern "C" void kernel_launch(void* const* d_in, const int* in_sizes, int n_in,
                              void* d_out, int out_size, void* d_ws, size_t ws_size,
                              hipStream_t stream) {
    const float* x    = (const float*)d_in[0];
    const float* tsq  = (const float*)d_in[1];
    const float* U_r  = (const float*)d_in[2];
    const float* W_r  = (const float*)d_in[3];
    const float* b_r  = (const float*)d_in[4];
    const float* U_c  = (const float*)d_in[5];
    const float* W_c  = (const float*)d_in[6];
    const float* b_c  = (const float*)d_in[7];
    const float* U_o  = (const float*)d_in[8];
    const float* W_o  = (const float*)d_in[9];
    const float* b_o  = (const float*)d_in[10];
    float* out    = (float*)d_out;                       // [B,T,H] hs | h_f | c_f
    float* out_hf = out + (size_t)BB * TT * HH;
    float* out_cf = out_hf + (size_t)BB * HH;

    // workspace carve (256B aligned)
    char* ws = (char*)d_ws;
    auto carve = [&](size_t bytes) {
        char* p = ws;
        ws += (bytes + 255) & ~(size_t)255;
        return p;
    };
    float*    xg   = (float*)carve(sizeof(float) * (size_t)TT * BB * H3);   // 384 MB
    __bf16*   x_bf = (__bf16*)carve(sizeof(__bf16) * (size_t)BB * TT * FF); // 32 MB
    __bf16*   Ut   = (__bf16*)carve(sizeof(__bf16) * (size_t)H3 * FF);      // 3 MB
    __bf16*   h_bf = (__bf16*)carve(sizeof(__bf16) * BB * HH);
    unsigned* cnt  = (unsigned*)carve(256);

    // allow >64KB dynamic LDS for the persistent kernel (non-stream host call)
    hipFuncSetAttribute(reinterpret_cast<const void*>(plstm_recurrence),
                        hipFuncAttributeMaxDynamicSharedMemorySize, LDS_BYTES);

    // 1) precision conversion + U packing (transposed, gate-concatenated)
    long nx = (long)BB * TT * FF;
    cast_f32_to_bf16<<<(nx + 255) / 256, 256, 0, stream>>>(x, x_bf, nx);
    int gU = (FF * HH + 255) / 256;
    transpose_cast<<<gU, 256, 0, stream>>>(U_r, Ut + 0 * (size_t)HH * FF, FF, HH);
    transpose_cast<<<gU, 256, 0, stream>>>(U_c, Ut + 1 * (size_t)HH * FF, FF, HH);
    transpose_cast<<<gU, 256, 0, stream>>>(U_o, Ut + 2 * (size_t)HH * FF, FF, HH);

    // 2) hoisted input projections: one WMMA GEMM for all 3 gates, time-major out
    {
        int waves = ((BB * TT) / 32) * (H3 / 64);    // 49152
        gemm_xproj<<<waves / 8, 256, 0, stream>>>(x_bf, Ut, xg);
    }

    // 3) recurrence: zero h state + barrier counter, then one persistent launch
    init_rec<<<(BB * HH + 255) / 256, 256, 0, stream>>>(h_bf, cnt);
    plstm_recurrence<<<NBLK, 256, LDS_BYTES, stream>>>(
        h_bf, xg, tsq, W_r, W_c, W_o, b_r, b_c, b_o, out, out_hf, out_cf, cnt);
}
